// TargetNet_68083821576665
// MI455X (gfx1250) — matrix operand order repaired and checked by reading, not compiled, not run on hardware
//
#include <hip/hip_runtime.h>
#include <hip/hip_bf16.h>

// ---------------------------------------------------------------------------
// Per-sample MLP (grouped 1x1 conv) for MI455X / gfx1250, bf16 WMMA path.
//   layer l:  Y[b, M, 256] = W[b, M, K] x X[b, K, 256] + bias, sigmoid
// Double-buffered LDS, K-stage = 64 (2 WMMA sub-steps per stage), async
// global->LDS copies for the bf16 activation tile, sched_group_barrier to
// batch fragment DS-reads ahead of the WMMA chain.
// ---------------------------------------------------------------------------

typedef __attribute__((ext_vector_type(16))) __bf16 v16bf;
typedef __attribute__((ext_vector_type(8)))  __bf16 v8bf;
typedef __attribute__((ext_vector_type(4)))  __bf16 v4bf;
typedef __attribute__((ext_vector_type(8)))  float  v8f;

// Pointer types matching the async-to-LDS builtin signature (from hipcc
// diagnostic: param0 is AS1 pointer to 'int __attribute__((vector_size(16)))').
typedef int v4i_vs __attribute__((vector_size(16)));
typedef __attribute__((address_space(1))) v4i_vs* as1_v4i_p;
typedef __attribute__((address_space(3))) v4i_vs* as3_v4i_p;

#define LDS_STRIDE 72   // 64 data halves + 8 pad -> 144B rows, 16B aligned, bank-friendly
#define KT 64           // K per LDS stage (2 WMMA k-steps of 32)

#if defined(__has_builtin)
#  if __has_builtin(__builtin_amdgcn_global_load_async_to_lds_b128)
#    define USE_ASYNC_LDS 1
#  endif
#  if __has_builtin(__builtin_amdgcn_sched_group_barrier)
#    define HAVE_SGB 1
#  endif
#endif
#ifndef USE_ASYNC_LDS
#  define USE_ASYNC_LDS 0
#endif
#ifndef HAVE_SGB
#  define HAVE_SGB 0
#endif

__device__ __forceinline__ void async_cp16(const __bf16* g, __bf16* l) {
#if USE_ASYNC_LDS
    __builtin_amdgcn_global_load_async_to_lds_b128(
        (as1_v4i_p)g, (as3_v4i_p)l, 0, 0);
#else
    (void)g; (void)l;
#endif
}

__device__ __forceinline__ void wait_async0() {
#if USE_ASYNC_LDS
#  if __has_builtin(__builtin_amdgcn_s_wait_asynccnt)
    __builtin_amdgcn_s_wait_asynccnt(0);
#  else
    asm volatile("s_wait_asynccnt 0" ::: "memory");
#  endif
#endif
}

// Request schedule: all fragment DS-reads first, then the 8 WMMAs.
__device__ __forceinline__ void shape_substep_schedule() {
#if HAVE_SGB
    __builtin_amdgcn_sched_group_barrier(0x100, 18, 0); // 18 x DS read (2 A + 16 B)
    __builtin_amdgcn_sched_group_barrier(0x008,  8, 0); // 8 x WMMA
#endif
}

// ---- direct staging helpers (prologue / non-pipelined paths) --------------

// W tile: 64 m-rows x 64 k, f32 -> bf16.  4 float4 per thread.
__device__ __forceinline__ void stageA_direct(__bf16* dst, const float* W,
                                              size_t rowBase, int K, int k0, int tid) {
#pragma unroll
    for (int p = 0; p < 4; ++p) {
        int q  = tid + p * 256;
        int m  = q >> 4;           // 16 float4 per 64-wide row
        int f4 = q & 15;
        float4 f = *(const float4*)(W + (rowBase + m) * K + k0 + f4 * 4);
        v4bf h;
        h[0] = (__bf16)f.x; h[1] = (__bf16)f.y; h[2] = (__bf16)f.z; h[3] = (__bf16)f.w;
        *(v4bf*)(&dst[m * LDS_STRIDE + f4 * 4]) = h;
    }
}

// Activation tile (layers>=2): 256 n-rows x 64 halves, straight 16B copies.
__device__ __forceinline__ void stageB_direct(__bf16* dst, const __bf16* actIn,
                                              size_t b, int K, int k0, int tid) {
#pragma unroll
    for (int p = 0; p < 8; ++p) {
        int q = tid + p * 256;
        int n = q >> 3;            // 8 x 16B chunks per 64-half row
        int c = q & 7;
        uint4 u = *(const uint4*)(actIn + (b * 256 + n) * K + k0 + c * 8);
        *(uint4*)(&dst[n * LDS_STRIDE + c * 8]) = u;
    }
}

// First-layer x tile: [k][256] f32, coalesced reads, transposed bf16 scatter.
__device__ __forceinline__ void stageB_first(__bf16* dst, const float* xIn,
                                             size_t b, int K, int k0, int tid) {
#pragma unroll
    for (int g = 0; g < 4; ++g) {
        float4 f[4];
#pragma unroll
        for (int p = 0; p < 4; ++p) {
            int q   = tid + (g * 4 + p) * 256;
            int kk  = q >> 6;              // 64 float4 per 256-wide row
            int nn4 = (q & 63) * 4;
            f[p] = *(const float4*)(xIn + (b * K + k0 + kk) * 256 + nn4);
        }
#pragma unroll
        for (int p = 0; p < 4; ++p) {
            int q   = tid + (g * 4 + p) * 256;
            int kk  = q >> 6;
            int nn4 = (q & 63) * 4;
            dst[(nn4 + 0) * LDS_STRIDE + kk] = (__bf16)f[p].x;
            dst[(nn4 + 1) * LDS_STRIDE + kk] = (__bf16)f[p].y;
            dst[(nn4 + 2) * LDS_STRIDE + kk] = (__bf16)f[p].z;
            dst[(nn4 + 3) * LDS_STRIDE + kk] = (__bf16)f[p].w;
        }
    }
}

template <bool FIRST>
__global__ __launch_bounds__(256)
void fused_fc_gemm(const float* __restrict__ W,     // [B][M][K] f32
                   const float* __restrict__ bias,  // [B][M]    f32
                   const float* __restrict__ xIn,   // FIRST : [B][K][256] f32 (k-major)
                   const __bf16* __restrict__ actIn,// !FIRST: [B][256][K] bf16 (n-major)
                   __bf16* __restrict__ actOut,     // [B][256][M] bf16
                   int M, int K)
{
    __shared__ __bf16 sA[2][64  * LDS_STRIDE];   // W tiles  (64 m x 64 k), bf16
    __shared__ __bf16 sB[2][256 * LDS_STRIDE];   // X tiles (256 n x 64 k), bf16

    const int    b       = blockIdx.y;
    const int    mBase   = blockIdx.x * 64;
    const size_t rowBase = (size_t)b * M + mBase;
    const int    tid     = threadIdx.x;
    const int    lane    = tid & 31;
    const int    wave    = tid >> 5;          // 8 waves
    const int    lm      = lane & 15;
    const int    hi      = lane >> 4;         // lane half-select
    const int    mSub    = wave & 3;          // 4 strips of 16 rows (M)
    const int    nSub    = wave >> 2;         // 2 halves of 128 cols (N)

    v8f acc[8];
#pragma unroll
    for (int j = 0; j < 8; ++j)
#pragma unroll
        for (int r = 0; r < 8; ++r) acc[j][r] = 0.0f;

    __bf16* curA = &sA[0][0]; __bf16* nxtA = &sA[1][0];
    __bf16* curB = &sB[0][0]; __bf16* nxtB = &sB[1][0];

    // ---- prologue: stage k0 = 0 ----
    stageA_direct(curA, W, rowBase, K, 0, tid);
    if constexpr (FIRST) stageB_first(curB, xIn, (size_t)b, K, 0, tid);
    else                 stageB_direct(curB, actIn, (size_t)b, K, 0, tid);
    __syncthreads();

    for (int k0 = 0; k0 < K; k0 += KT) {
        const int  kn      = k0 + KT;
        const bool hasNext = kn < K;

        // ---- issue next-stage global loads BEFORE the WMMA chain ----
        float4 wreg[4];
#if !USE_ASYNC_LDS
        uint4 breg[8];
#endif
        if (hasNext) {
#pragma unroll
            for (int p = 0; p < 4; ++p) {
                int q  = tid + p * 256;
                int m  = q >> 4;
                int f4 = q & 15;
                wreg[p] = *(const float4*)(W + (rowBase + m) * K + kn + f4 * 4);
            }
            if constexpr (!FIRST) {
#if USE_ASYNC_LDS
#pragma unroll
                for (int p = 0; p < 8; ++p) {
                    int q = tid + p * 256;
                    int n = q >> 3;
                    int c = q & 7;
                    async_cp16(actIn + ((size_t)b * 256 + n) * K + kn + c * 8,
                               nxtB + n * LDS_STRIDE + c * 8);
                }
#else
#pragma unroll
                for (int p = 0; p < 8; ++p) {
                    int q = tid + p * 256;
                    int n = q >> 3;
                    int c = q & 7;
                    breg[p] = *(const uint4*)(actIn + ((size_t)b * 256 + n) * K + kn + c * 8);
                }
#endif
            }
        }

        // ---- compute: 2 k-substeps x 8 WMMAs, fragments preloaded ----
#pragma unroll
        for (int s = 0; s < 2; ++s) {
            // A fragment (16x32 bf16, wave32 layout):
            //   elements 0..7 = k s*32+hi*8+0..7 ; 8..15 = k s*32+hi*8+16..23
            const __bf16* pa = &curA[(mSub * 16 + lm) * LDS_STRIDE + s * 32 + hi * 8];
            v8bf aLo = *(const v8bf*)(pa);
            v8bf aHi = *(const v8bf*)(pa + 16);
            v16bf a  = __builtin_shufflevector(aLo, aHi, 0, 1, 2, 3, 4, 5, 6, 7,
                                                         8, 9, 10, 11, 12, 13, 14, 15);
            v16bf bb[8];
#pragma unroll
            for (int j = 0; j < 8; ++j) {
                // B fragment (32x16): n = lm, k = s*32 + hi*8 + 0..7 ; s*32 + hi*8 + 16..23
                const __bf16* pb =
                    &curB[(nSub * 128 + j * 16 + lm) * LDS_STRIDE + s * 32 + hi * 8];
                v8bf bLo = *(const v8bf*)(pb);
                v8bf bHi = *(const v8bf*)(pb + 16);
                bb[j] = __builtin_shufflevector(bLo, bHi, 0, 1, 2, 3, 4, 5, 6, 7,
                                                          8, 9, 10, 11, 12, 13, 14, 15);
            }
#pragma unroll
            for (int j = 0; j < 8; ++j)
                acc[j] = __builtin_amdgcn_wmma_f32_16x16x32_bf16(
                    false, a, false, bb[j], (short)0, acc[j], false, false);
            shape_substep_schedule();   // [18 x ds_read] then [8 x wmma]
        }

        // ---- commit next-stage data to LDS AFTER the WMMA chain ----
        if (hasNext) {
#pragma unroll
            for (int p = 0; p < 4; ++p) {
                int q  = tid + p * 256;
                int m  = q >> 4;
                int f4 = q & 15;
                v4bf h;
                h[0] = (__bf16)wreg[p].x; h[1] = (__bf16)wreg[p].y;
                h[2] = (__bf16)wreg[p].z; h[3] = (__bf16)wreg[p].w;
                *(v4bf*)(&nxtA[m * LDS_STRIDE + f4 * 4]) = h;
            }
            if constexpr (FIRST) {
                stageB_first(nxtB, xIn, (size_t)b, K, kn, tid);
            } else {
#if USE_ASYNC_LDS
                wait_async0();
#else
#pragma unroll
                for (int p = 0; p < 8; ++p) {
                    int q = tid + p * 256;
                    int n = q >> 3;
                    int c = q & 7;
                    *(uint4*)(&nxtB[n * LDS_STRIDE + c * 8]) = breg[p];
                }
#endif
            }
        }
        __syncthreads();
        { __bf16* t = curA; curA = nxtA; nxtA = t; }
        { __bf16* t = curB; curB = nxtB; nxtB = t; }
    }

    // ---- epilogue: bias + sigmoid, bf16 pack, 16B store per tile per lane ----
    const int mLoc = mSub * 16 + hi * 8;   // C layout: m = hi*8 + vgpr_index
    float bv[8];
#pragma unroll
    for (int r = 0; r < 8; ++r)
        bv[r] = bias[(size_t)b * M + mBase + mLoc + r];

#pragma unroll
    for (int j = 0; j < 8; ++j) {
        int n = nSub * 128 + j * 16 + lm;  // C layout: n = lane & 15 within tile
        v8bf outv;
#pragma unroll
        for (int r = 0; r < 8; ++r) {
            float t = acc[j][r] + bv[r];
            float s = 1.0f / (1.0f + __expf(-t));
            outv[r] = (__bf16)s;
        }
        *(v8bf*)(&actOut[((size_t)b * 256 + n) * M + mBase + mLoc]) = outv;
    }
}

// Final layer: M = 1, K = 128 -> per-(b,n) 64-length dot product (k 0..63), no sigmoid.
__global__ __launch_bounds__(256)
void fc_final(const float* __restrict__ W,      // [B][1][128]
              const float* __restrict__ bias,   // [B]
              const __bf16* __restrict__ actIn, // [B][256][128]
              float* __restrict__ out)          // [B][256]
{
    int b = blockIdx.x;
    int n = threadIdx.x;
    const __bf16* row = actIn + ((size_t)b * 256 + n) * 128;
    const float*  w   = W + (size_t)b * 128;
    float acc = 0.0f;
#pragma unroll
    for (int c = 0; c < 8; ++c) {
        v8bf h = *(const v8bf*)(row + c * 8);   // 16B vector load
#pragma unroll
        for (int r = 0; r < 8; ++r)
            acc += (float)h[r] * w[c * 8 + r];
    }
    out[(size_t)b * 256 + n] = acc + bias[b];
}

extern "C" void kernel_launch(void* const* d_in, const int* in_sizes, int n_in,
                              void* d_out, int out_size, void* d_ws, size_t ws_size,
                              hipStream_t stream)
{
    (void)in_sizes; (void)n_in; (void)out_size; (void)ws_size;
    const float* x  = (const float*)d_in[0];
    const float* w1 = (const float*)d_in[1]; const float* b1 = (const float*)d_in[2];
    const float* w2 = (const float*)d_in[3]; const float* b2 = (const float*)d_in[4];
    const float* w3 = (const float*)d_in[5]; const float* b3 = (const float*)d_in[6];
    const float* w4 = (const float*)d_in[7]; const float* b4 = (const float*)d_in[8];
    const float* w5 = (const float*)d_in[9]; const float* b5 = (const float*)d_in[10];

    // workspace: act1 (bf16 64*256*1024 = 32MB) | act2 (bf16 64*256*512 = 16MB)
    // act3 reuses act1 region (8MB), act4 reuses act2 region (4MB). Max = 48MB.
    __bf16* act1 = (__bf16*)d_ws;
    __bf16* act2 = (__bf16*)((char*)d_ws + (size_t)64 * 256 * 1024 * 2);
    __bf16* act3 = act1;
    __bf16* act4 = act2;
    float*  out  = (float*)d_out;

    dim3 blk(256);
    fused_fc_gemm<true ><<<dim3(16, 64), blk, 0, stream>>>(w1, b1, x,       nullptr, act1, 1024, 2048);
    fused_fc_gemm<false><<<dim3( 8, 64), blk, 0, stream>>>(w2, b2, nullptr, act1,    act2,  512, 1024);
    fused_fc_gemm<false><<<dim3( 4, 64), blk, 0, stream>>>(w3, b3, nullptr, act2,    act3,  256,  512);
    fused_fc_gemm<false><<<dim3( 2, 64), blk, 0, stream>>>(w4, b4, nullptr, act3,    act4,  128,  256);
    fc_final<<<dim3(64), blk, 0, stream>>>(w5, b5, act4, out);
}
